// LightAttention_22917945491886
// MI455X (gfx1250) — matrix operand order repaired and checked by reading, not compiled, not run on hardware
//
#include <hip/hip_runtime.h>
#include <hip/hip_bf16.h>
#include <cstdint>

#define B_ 8
#define C_ 16
#define S_ 512
#define N_ 512
#define V_ 128

typedef __bf16 bf16_t;
typedef __attribute__((ext_vector_type(16))) __bf16 v16bf;
typedef __attribute__((ext_vector_type(8)))  __bf16 v8bf;
typedef __attribute__((ext_vector_type(8)))  float  v8f;

__device__ __forceinline__ v8f zero8() {
  v8f z = {0.f,0.f,0.f,0.f,0.f,0.f,0.f,0.f};
  return z;
}

__device__ __forceinline__ v16bf cat16(v8bf lo, v8bf hi) {
  return __builtin_shufflevector(lo, hi, 0,1,2,3,4,5,6,7,8,9,10,11,12,13,14,15);
}

// A fragment (16x32 bf16, row-major source), base pre-offset to (row0, k0).
// ISA layout: lane L holds row m=L&15; K {kb..kb+7} in elems 0..7 and
// {kb+16..kb+23} in elems 8..15, kb = (L<16)?0:8.
__device__ __forceinline__ v16bf load_a_frag(const bf16_t* A, int lda, int lane) {
  int m  = lane & 15;
  int kb = (lane & 16) ? 8 : 0;
  const bf16_t* p = A + (long)m * lda + kb;
  v8bf lo = *(const v8bf*)(p);
  v8bf hi = *(const v8bf*)(p + 16);
  return cat16(lo, hi);
}

// B fragment (32x16 bf16) loaded from B-TRANSPOSED row-major [n, K].
// ISA layout: lane L holds col n=L&15; K {kb..kb+7} in elems 0..7 and
// {kb+16..kb+23} in elems 8..15, kb = (L<16)?0:8.
__device__ __forceinline__ v16bf load_bt_frag(const bf16_t* BT, int ldbt, int lane) {
  int n  = lane & 15;
  int kb = (lane & 16) ? 8 : 0;
  const bf16_t* p = BT + (long)n * ldbt + kb;
  v8bf lo = *(const v8bf*)(p);
  v8bf hi = *(const v8bf*)(p + 16);
  return cat16(lo, hi);
}

__device__ __forceinline__ v8f wmma_bf16(v16bf a, v16bf b, v8f c) {
  return __builtin_amdgcn_wmma_f32_16x16x32_bf16(false, a, false, b, (short)0, c,
                                                 false, false);
}

// C/D layout: VGPR r, lane L -> (m = r + (L<16?0:8), n = L&15)
__device__ __forceinline__ void store_c_bf16(bf16_t* Cp, int ldc, int lane, v8f acc) {
  int n  = lane & 15;
  int mb = (lane & 16) ? 8 : 0;
#pragma unroll
  for (int r = 0; r < 8; ++r) Cp[(long)(mb + r) * ldc + n] = (bf16_t)acc[r];
}

__device__ __forceinline__ void store_c_bf16_T(bf16_t* CTp, int ldct, int lane, v8f acc) {
  int n  = lane & 15;
  int mb = (lane & 16) ? 8 : 0;
#pragma unroll
  for (int r = 0; r < 8; ++r) CTp[(long)n * ldct + (mb + r)] = (bf16_t)acc[r];
}

// ---------------- Kernel 1: x -> bf16 (row-major + transposed) ----------------
__global__ __launch_bounds__(256) void k_conv_x(const float* __restrict__ x,
                                                bf16_t* __restrict__ xb,
                                                bf16_t* __restrict__ xbT) {
  __shared__ __align__(32) bf16_t tile[32][33];
  int bc = blockIdx.z;
  int s0 = blockIdx.x * 32, n0 = blockIdx.y * 32;
  const float* xp = x + (long)bc * S_ * N_;
  bf16_t* xbp = xb + (long)bc * S_ * N_;
#pragma unroll
  for (int r = threadIdx.y; r < 32; r += 8) {
    float v = xp[(long)(s0 + r) * N_ + n0 + threadIdx.x];
    bf16_t bv = (bf16_t)v;
    xbp[(long)(s0 + r) * N_ + n0 + threadIdx.x] = bv;
    tile[threadIdx.x][r] = bv;   // transposed staging
  }
  __syncthreads();
  bf16_t* tp = xbT + (long)bc * N_ * S_;
#pragma unroll
  for (int r = threadIdx.y; r < 32; r += 8) {
    tp[(long)(n0 + r) * S_ + s0 + threadIdx.x] = tile[r][threadIdx.x];
  }
}

// ---------------- Kernel 2: weights -> bf16 layouts, zero accumulators -------
__global__ __launch_bounds__(256) void k_conv_w(
    const float* __restrict__ wq, const float* __restrict__ k_left,
    const float* __restrict__ k_right, const float* __restrict__ v1_left,
    const float* __restrict__ v1_right, const float* __restrict__ v2_w,
    bf16_t* __restrict__ Wleft, bf16_t* __restrict__ wqT,
    bf16_t* __restrict__ krT, bf16_t* __restrict__ v1rT,
    float* __restrict__ accum) {
  int i = blockIdx.x * 256 + threadIdx.x;
  if (i < 2 * C_) accum[i] = 0.f;
  const int WL = C_ * 3 * V_ * S_;   // stacked left weights
  const int WT = C_ * V_ * N_;       // each transposed right weight
  if (i < WL) {
    int c   = i / (3 * V_ * S_);
    int rem = i % (3 * V_ * S_);
    int r   = rem / S_;
    int s   = rem % S_;
    float v;
    if (r < V_)           v = k_left [((long)c * V_ + r)        * S_ + s];
    else if (r < 2 * V_)  v = v1_left[((long)c * V_ + (r - V_)) * S_ + s];
    else                  v = v2_w  [((long)c * V_ + (r - 2*V_))* S_ + s];
    Wleft[i] = (bf16_t)v;
  } else if (i < WL + 3 * WT) {
    int j     = i - WL;
    int which = j / WT;
    int jj    = j % WT;
    int c   = jj / (V_ * N_);
    int rem = jj % (V_ * N_);
    int v   = rem / N_;
    int n   = rem % N_;
    const float* src = (which == 0) ? wq : (which == 1) ? k_right : v1_right;
    bf16_t*      dst = (which == 0) ? wqT : (which == 1) ? krT : v1rT;
    dst[jj] = (bf16_t)src[((long)c * N_ + n) * V_ + v];  // [C,N,V] -> [C,V,N]
  }
}

// ---------------- Kernel 3: P = [k_left; v1_left; v2_w] @ x  (M=384,K=512) ---
// 32x32 C-tile per wave (2x2 WMMA register blocking).
// rows 0..255 -> Pa row-major [256,N]; rows 256..383 -> Pv2T transposed [N,V]
__global__ __launch_bounds__(256) void k_gemm_left(
    const bf16_t* __restrict__ Wleft, const bf16_t* __restrict__ xbT,
    bf16_t* __restrict__ Pa, bf16_t* __restrict__ Pv2T) {
  int gw   = (blockIdx.x * 256 + threadIdx.x) >> 5;
  int lane = threadIdx.x & 31;
  const int TPB = 12 * 16;          // (384/32)*(512/32) tiles per (b,c)
  int bc = gw / TPB;
  int t  = gw % TPB;
  int tm = t / 16, tn = t % 16;
  int c  = bc % C_;
  const bf16_t* Ab = Wleft + ((long)c * 3 * V_ + tm * 32) * S_;
  const bf16_t* Bb = xbT + ((long)bc * N_ + tn * 32) * S_;
  v8f a00 = zero8(), a01 = zero8(), a10 = zero8(), a11 = zero8();
#pragma unroll 2
  for (int k = 0; k < S_; k += 32) {
    __builtin_prefetch(Ab + k + 128, 0, 1);
    __builtin_prefetch(Bb + k + 128, 0, 1);
    v16bf fa0 = load_a_frag(Ab + k, S_, lane);
    v16bf fa1 = load_a_frag(Ab + 16 * S_ + k, S_, lane);
    v16bf fb0 = load_bt_frag(Bb + k, S_, lane);
    v16bf fb1 = load_bt_frag(Bb + 16 * S_ + k, S_, lane);
    a00 = wmma_bf16(fa0, fb0, a00);
    a01 = wmma_bf16(fa0, fb1, a01);
    a10 = wmma_bf16(fa1, fb0, a10);
    a11 = wmma_bf16(fa1, fb1, a11);
  }
  int row0 = tm * 32, col0 = tn * 32;
  if (row0 < 2 * V_) {
    bf16_t* base = Pa + (long)bc * (2 * V_) * N_;
    store_c_bf16(base + (long)row0 * N_ + col0,             N_, lane, a00);
    store_c_bf16(base + (long)row0 * N_ + col0 + 16,        N_, lane, a01);
    store_c_bf16(base + (long)(row0 + 16) * N_ + col0,      N_, lane, a10);
    store_c_bf16(base + (long)(row0 + 16) * N_ + col0 + 16, N_, lane, a11);
  } else {
    bf16_t* base = Pv2T + (long)bc * N_ * V_ + (row0 - 2 * V_);
    store_c_bf16_T(base + (long)col0 * V_,             V_, lane, a00);
    store_c_bf16_T(base + (long)(col0 + 16) * V_,      V_, lane, a01);
    store_c_bf16_T(base + (long)col0 * V_ + 16,        V_, lane, a10);
    store_c_bf16_T(base + (long)(col0 + 16) * V_ + 16, V_, lane, a11);
  }
}

// ---------------- Kernel 4: k = Pk@k_right, v1 = Pv1@v1_right (V x V) --------
__global__ __launch_bounds__(256) void k_gemm_kv1(
    const bf16_t* __restrict__ Pa, const bf16_t* __restrict__ krT,
    const bf16_t* __restrict__ v1rT, bf16_t* __restrict__ kmat,
    bf16_t* __restrict__ v1mat) {
  int gw   = (blockIdx.x * 256 + threadIdx.x) >> 5;
  int lane = threadIdx.x & 31;
  const int TPB = 16 * 8;           // (256/16)*(128/16)
  int bc = gw / TPB;
  int t  = gw % TPB;
  int tm = t / 8, tn = t % 8;
  int c  = bc % C_;
  const bf16_t* Ab = Pa + ((long)bc * 2 * V_ + tm * 16) * N_;
  const bf16_t* BTw = (tm < 8) ? krT : v1rT;
  const bf16_t* Bb = BTw + ((long)c * V_ + tn * 16) * N_;
  v8f acc = zero8();
#pragma unroll 4
  for (int k = 0; k < N_; k += 32) {
    acc = wmma_bf16(load_a_frag(Ab + k, N_, lane), load_bt_frag(Bb + k, N_, lane), acc);
  }
  bf16_t* out = (tm < 8)
      ? (kmat  + (long)bc * V_ * V_ + (long)(tm * 16) * V_ + tn * 16)
      : (v1mat + (long)bc * V_ * V_ + (long)((tm - 8) * 16) * V_ + tn * 16);
  store_c_bf16(out, V_, lane, acc);
}

// ---------------- Kernel 5: vT = v2^T @ v1^T  ([N,V] = [N,V]@..., BT=v1) -----
__global__ __launch_bounds__(256) void k_gemm_vt(
    const bf16_t* __restrict__ Pv2T, const bf16_t* __restrict__ v1mat,
    bf16_t* __restrict__ vT) {
  int gw   = (blockIdx.x * 256 + threadIdx.x) >> 5;
  int lane = threadIdx.x & 31;
  const int TPB = 32 * 8;           // (512/16)*(128/16)
  int bc = gw / TPB;
  int t  = gw % TPB;
  int tm = t / 8, tn = t % 8;
  const bf16_t* Ab = Pv2T + ((long)bc * N_ + tm * 16) * V_;
  const bf16_t* Bb = v1mat + ((long)bc * V_ + tn * 16) * V_;
  v8f acc = zero8();
#pragma unroll
  for (int k = 0; k < V_; k += 32) {
    acc = wmma_bf16(load_a_frag(Ab + k, V_, lane), load_bt_frag(Bb + k, V_, lane), acc);
  }
  store_c_bf16(vT + (long)bc * N_ * V_ + (long)(tm * 16) * V_ + tn * 16, V_, lane, acc);
}

// ---------------- Kernel 6: fused q -> logits -> softmax -> attn@v + residual
// kmat tile (32KB) is DMA'd to LDS with global_load_async_to_lds_b128, issued
// BEFORE the q GEMM so the async copy overlaps the q WMMAs (ASYNCcnt drain
// right before the barrier).
__global__ __launch_bounds__(128) void k_attn(
    const bf16_t* __restrict__ xb, const bf16_t* __restrict__ wqT,
    const bf16_t* __restrict__ kmat, const bf16_t* __restrict__ vT,
    const float* __restrict__ x, float* __restrict__ y,
    float* __restrict__ accum) {
  __shared__ __align__(32) bf16_t smem[4][16 * 128];
  __shared__ __align__(16) bf16_t ldsK[V_ * V_];   // 32KB, shared by the block
  int wid = threadIdx.x >> 5, lane = threadIdx.x & 31;
  int gw = blockIdx.x * 4 + wid;
  int bc = gw / (S_ / 16);       // all 4 waves of a block share bc (32 % 4 == 0)
  int st = gw % (S_ / 16);
  int c  = bc % C_;
  int s0 = st * 16;
  bf16_t* lq = smem[wid];

  // ---- issue async global->LDS copy of kmat[bc] (2048 x b128, 16/thread) ----
  {
    const bf16_t* Kg = kmat + (long)bc * V_ * V_;
    int t = threadIdx.x;
#pragma unroll
    for (int i = 0; i < 16; ++i) {
      int e = (t + i * 128) * 8;   // element offset, 8 bf16 = 16 bytes
      unsigned long long ga = (unsigned long long)(uintptr_t)(Kg + e);
      unsigned int la = (unsigned int)(uintptr_t)(&ldsK[e]);
      asm volatile("global_load_async_to_lds_b128 %0, %1, off"
                   :: "v"(la), "v"(ga) : "memory");
    }
  }

  // ---- q = x[s0:s0+16,:] @ wq  ([16, V] in 8 accumulator tiles) ----
  const bf16_t* Ab = xb + ((long)bc * S_ + s0) * N_;
  const bf16_t* WQ = wqT + (long)c * V_ * N_;
  v8f qa[8];
#pragma unroll
  for (int j = 0; j < 8; ++j) qa[j] = zero8();
  for (int k = 0; k < N_; k += 32) {
    v16bf a = load_a_frag(Ab + k, N_, lane);
#pragma unroll
    for (int j = 0; j < 8; ++j)
      qa[j] = wmma_bf16(a, load_bt_frag(WQ + (long)(j * 16) * N_ + k, N_, lane), qa[j]);
  }
  // stage q (bf16) to LDS row-major [16,128] to re-layout as A fragments
  {
    int n = lane & 15, mb = (lane & 16) ? 8 : 0;
#pragma unroll
    for (int j = 0; j < 8; ++j)
#pragma unroll
      for (int r = 0; r < 8; ++r)
        lq[(mb + r) * 128 + j * 16 + n] = (bf16_t)qa[j][r];
  }
  asm volatile("s_wait_asynccnt 0x0" ::: "memory");
  __syncthreads();

  // ---- logits = q @ k^T  (BT = kmat row-major [V,V], now in LDS) ----
  v8f la[8];
#pragma unroll
  for (int j = 0; j < 8; ++j) la[j] = zero8();
#pragma unroll
  for (int kc = 0; kc < 4; ++kc) {
    v16bf a = load_a_frag(lq + kc * 32, 128, lane);
#pragma unroll
    for (int j = 0; j < 8; ++j)
      la[j] = wmma_bf16(a, load_bt_frag(ldsK + (long)(j * 16) * V_ + kc * 32, V_, lane),
                        la[j]);
  }

  // ---- softmax over 128 cols per row (halves of the wave are independent) --
  float rmax[8], rsum[8];
#pragma unroll
  for (int r = 0; r < 8; ++r) {
    float m = la[0][r];
#pragma unroll
    for (int j = 1; j < 8; ++j) m = fmaxf(m, la[j][r]);
#pragma unroll
    for (int d = 1; d < 16; d <<= 1) m = fmaxf(m, __shfl_xor(m, d, 32));
    rmax[r] = m;
  }
#pragma unroll
  for (int r = 0; r < 8; ++r) {
    float s = 0.f;
#pragma unroll
    for (int j = 0; j < 8; ++j) {
      float e = __expf(la[j][r] - rmax[r]);
      la[j][r] = e;
      s += e;
    }
#pragma unroll
    for (int d = 1; d < 16; d <<= 1) s += __shfl_xor(s, d, 32);
    rsum[r] = s;
  }
  __syncthreads();
  {
    int n = lane & 15, mb = (lane & 16) ? 8 : 0;
#pragma unroll
    for (int j = 0; j < 8; ++j)
#pragma unroll
      for (int r = 0; r < 8; ++r)
        lq[(mb + r) * 128 + j * 16 + n] = (bf16_t)(la[j][r] / rsum[r]);
  }
  __syncthreads();
  v16bf af[4];
#pragma unroll
  for (int kc = 0; kc < 4; ++kc) af[kc] = load_a_frag(lq + kc * 32, 128, lane);

  // ---- y = x + attn @ v  (BT = vT row-major [N,V]), 128-col chunks ----
  const bf16_t* VTb = vT + (long)bc * N_ * V_;
  const float*  Xr  = x + ((long)bc * S_ + s0) * N_;
  float*        Yr  = y + ((long)bc * S_ + s0) * N_;
  float lsum = 0.f, lsq = 0.f;
  int n = lane & 15, mb = (lane & 16) ? 8 : 0;
#pragma unroll
  for (int nc = 0; nc < 4; ++nc) {
    v8f ya[8];
#pragma unroll
    for (int j = 0; j < 8; ++j) ya[j] = zero8();
#pragma unroll
    for (int j = 0; j < 8; ++j) {
      const bf16_t* bt = VTb + (long)(nc * 128 + j * 16) * V_;
#pragma unroll
      for (int kc = 0; kc < 4; ++kc)
        ya[j] = wmma_bf16(af[kc], load_bt_frag(bt + kc * 32, V_, lane), ya[j]);
    }
#pragma unroll
    for (int j = 0; j < 8; ++j)
#pragma unroll
      for (int r = 0; r < 8; ++r) {
        long off = (long)(mb + r) * N_ + nc * 128 + j * 16 + n;
        float yv = ya[j][r] + Xr[off];
        Yr[off] = yv;
        lsum += yv;
        lsq  += yv * yv;
      }
  }
#pragma unroll
  for (int d = 1; d < 32; d <<= 1) {
    lsum += __shfl_xor(lsum, d, 32);
    lsq  += __shfl_xor(lsq,  d, 32);
  }
  if (lane == 0) {
    atomicAdd(&accum[c], lsum);
    atomicAdd(&accum[C_ + c], lsq);
  }
}

// ---------------- Kernel 7: batch-scaler (in place on d_out) -----------------
__global__ __launch_bounds__(256) void k_norm(float* __restrict__ y,
                                              const float* __restrict__ accum,
                                              const float* __restrict__ gamma,
                                              const float* __restrict__ beta) {
  long i = (long)blockIdx.x * 256 + threadIdx.x;
  const long total = (long)B_ * C_ * S_ * N_;
  if (i >= total) return;
  int c = (int)((i / ((long)S_ * N_)) % C_);
  const float cnt = (float)((long)B_ * S_ * N_);
  float mean = accum[c] / cnt;
  float var  = accum[C_ + c] / cnt - mean * mean;
  float rstd = rsqrtf(var + 1e-5f);
  y[i] = (y[i] - mean) * rstd * gamma[c] + beta[c];
}

extern "C" void kernel_launch(void* const* d_in, const int* in_sizes, int n_in,
                              void* d_out, int out_size, void* d_ws, size_t ws_size,
                              hipStream_t stream) {
  const float* x        = (const float*)d_in[0];
  const float* wq       = (const float*)d_in[1];
  const float* k_left   = (const float*)d_in[2];
  const float* k_right  = (const float*)d_in[3];
  const float* v1_left  = (const float*)d_in[4];
  const float* v1_right = (const float*)d_in[5];
  const float* v2_w     = (const float*)d_in[6];
  const float* gamma    = (const float*)d_in[7];
  const float* beta     = (const float*)d_in[8];
  float* y = (float*)d_out;

  uintptr_t w = (uintptr_t)d_ws;
  auto alloc = [&](size_t bytes) {
    uintptr_t p = (w + 255) & ~(uintptr_t)255;
    w = p + bytes;
    return (void*)p;
  };
  const long XSZ = (long)B_ * C_ * S_ * N_;
  bf16_t* xb    = (bf16_t*)alloc(XSZ * 2);
  bf16_t* xbT   = (bf16_t*)alloc(XSZ * 2);
  bf16_t* Wleft = (bf16_t*)alloc((long)C_ * 3 * V_ * S_ * 2);
  bf16_t* wqT   = (bf16_t*)alloc((long)C_ * V_ * N_ * 2);
  bf16_t* krT   = (bf16_t*)alloc((long)C_ * V_ * N_ * 2);
  bf16_t* v1rT  = (bf16_t*)alloc((long)C_ * V_ * N_ * 2);
  bf16_t* Pa    = (bf16_t*)alloc((long)B_ * C_ * 2 * V_ * N_ * 2);
  bf16_t* Pv2T  = (bf16_t*)alloc((long)B_ * C_ * N_ * V_ * 2);
  bf16_t* kmat  = (bf16_t*)alloc((long)B_ * C_ * V_ * V_ * 2);
  bf16_t* v1mat = (bf16_t*)alloc((long)B_ * C_ * V_ * V_ * 2);
  bf16_t* vT    = (bf16_t*)alloc((long)B_ * C_ * N_ * V_ * 2);
  float*  accum = (float*)alloc(2 * C_ * sizeof(float));

  // 1) conversions
  k_conv_x<<<dim3(S_ / 32, N_ / 32, B_ * C_), dim3(32, 8), 0, stream>>>(x, xb, xbT);
  {
    int tot = C_ * 3 * V_ * S_ + 3 * C_ * V_ * N_;
    k_conv_w<<<(tot + 255) / 256, 256, 0, stream>>>(wq, k_left, k_right, v1_left,
                                                    v1_right, v2_w, Wleft, wqT, krT,
                                                    v1rT, accum);
  }
  // 2) P = W_left @ x  (32x32 tile per wave)
  {
    int waves = B_ * C_ * 12 * 16;
    k_gemm_left<<<waves / 8, 256, 0, stream>>>(Wleft, xbT, Pa, Pv2T);
  }
  // 3) k, v1
  {
    int waves = B_ * C_ * 16 * 8;
    k_gemm_kv1<<<waves / 8, 256, 0, stream>>>(Pa, krT, v1rT, kmat, v1mat);
  }
  // 4) vT = v2^T @ v1^T
  {
    int waves = B_ * C_ * 32 * 8;
    k_gemm_vt<<<waves / 8, 256, 0, stream>>>(Pv2T, v1mat, vT);
  }
  // 5) fused attention + residual + BN partial sums
  {
    int waves = B_ * C_ * (S_ / 16);
    k_attn<<<waves / 4, 128, 0, stream>>>(xb, wqT, kmat, vT, x, y, accum);
  }
  // 6) batch-scaler
  {
    long tot = XSZ;
    k_norm<<<(int)((tot + 255) / 256), 256, 0, stream>>>(y, accum, gamma, beta);
  }
}
